// SplatInformationFlow_38431367364863
// MI455X (gfx1250) — matrix operand order repaired and checked by reading, not compiled, not run on hardware
//
#include <hip/hip_runtime.h>
#include <hip/hip_bf16.h>
#include <stdint.h>

// ---------------------------------------------------------------------------
// SplatInformationFlow on MI455X (gfx1250).
// GEMMs via v_wmma_f32_16x16x32_bf16 (fp32 accum); LDS tiles staged by the
// Tensor Data Mover (tensor_load_to_lds + s_wait_tensorcnt) where possible.
// WG = 256 threads = 8 wave32s; WG tile 64x128; wave tile 32x32 (4 WMMA/K-step);
// double-buffered LDS; templated stage/epilogue (no runtime dispatch).
// ---------------------------------------------------------------------------

typedef __attribute__((ext_vector_type(16))) __bf16    v16bf;
typedef __attribute__((ext_vector_type(8)))  float     v8f;
typedef __attribute__((ext_vector_type(4)))  uint32_t  v4u;
typedef __attribute__((ext_vector_type(8)))  int       v8i;
typedef __attribute__((ext_vector_type(4)))  int       v4i;

#define BM 64
#define BN 128
#define BK 32
#define NT 256

static __device__ __forceinline__ uint16_t f2bf(float f) {
  union { float f; uint32_t u; } x; x.f = f;
  uint32_t r = x.u + 0x7FFFu + ((x.u >> 16) & 1u);   // round-to-nearest-even
  return (uint16_t)(r >> 16);
}

// TDM: DMA a 2D bf16 tile (tile_d0 contiguous elems x tile_d1 rows, row stride
// `stride_elems`) from global to LDS at byte offset lds_off (rows packed
// contiguously -> LDS layout [row][tile_d0]).  All inputs are wave-uniform.
static __device__ __forceinline__ void tdm_load_2d(uint32_t lds_off,
                                                   const uint16_t* gptr,
                                                   uint32_t tile_d0, uint32_t tile_d1,
                                                   uint32_t stride_elems) {
  uint64_t ga = (uint64_t)(uintptr_t)gptr;
  v4u g0;
  g0[0] = 1u;                                        // count=1, user descriptor
  g0[1] = lds_off;                                   // lds_addr (bytes)
  g0[2] = (uint32_t)ga;                              // global_addr[31:0]
  g0[3] = (uint32_t)((ga >> 32) & 0x01FFFFFFu) | (2u << 30);   // addr[56:32] | type=2
  v8i g1;
  g1[0] = (int)(1u << 16);                           // data_size = 2 bytes; mask=0
  g1[1] = (int)((tile_d0 & 0xFFFFu) << 16);          // tensor_dim0[15:0]
  g1[2] = (int)(((tile_d0 >> 16) & 0xFFFFu) |        // tensor_dim0[31:16]
                ((tile_d1 & 0xFFFFu) << 16));        // tensor_dim1[15:0]
  g1[3] = (int)(((tile_d1 >> 16) & 0xFFFFu) |        // tensor_dim1[31:16]
                ((tile_d0 & 0xFFFFu) << 16));        // tile_dim0
  g1[4] = (int)(tile_d1 & 0xFFFFu);                  // tile_dim1 ; tile_dim2 = 0
  g1[5] = (int)stride_elems;                         // tensor_dim0_stride[31:0]
  g1[6] = 0;                                         // stride[47:32]; dim1_stride lo
  g1[7] = 0;
  v4i z4 = {0, 0, 0, 0};
#if defined(__clang_major__) && (__clang_major__ >= 23)
  v8i z8 = {0, 0, 0, 0, 0, 0, 0, 0};
  __builtin_amdgcn_tensor_load_to_lds(g0, g1, z4, z4, z8, 0);
#else
  __builtin_amdgcn_tensor_load_to_lds(g0, g1, z4, z4, 0);
#endif
}

// 16x32 bf16 WMMA fragment from an LDS tile with row stride BK u16.
// A (rows = M): lane(r=L&15,h=L>>4): VGPR j<4 -> K=2j+8h ; j>=4 -> K=16+2(j-4)+8h
//   => two 16B ds loads at u16 offsets (r*BK + 8h) and (+16).
// B staged transposed [n][k] (rows = N): same K mapping as A
//   => two 16B ds loads at u16 offsets (r*BK + 8h) and (+16).
static __device__ __forceinline__ v16bf load_frag(const uint16_t* tile, int lane, bool isA) {
  int r = lane & 15;
  int h = lane >> 4;
  const uint16_t* p0 = isA ? (tile + r * BK + h * 8)  : (tile + r * BK + h * 8);
  const uint16_t* p1 = isA ? (p0 + 16)                : (p0 + 16);
  union { uint4 q[2]; v16bf v; } f;
  f.q[0] = *reinterpret_cast<const uint4*>(p0);
  f.q[1] = *reinterpret_cast<const uint4*>(p1);
  return f.v;
}

static __device__ __forceinline__ v8f wmma_bf16(v16bf a, v16bf b, v8f c) {
  return __builtin_amdgcn_wmma_f32_16x16x32_bf16(false, a, false, b, (short)0, c,
                                                 false, false);
}

// AMODE: 0 = A from f32 (convert at stage; EMODE==1 also applies per-row scale)
//        1 = A from bf16 row-major  -> staged by TDM
//        2 = A from bf16 with A(m,k) = src[k*lda + m] (gather; M==BM, m0==0)
// EMODE: 0 = store f32
//        1 = +bias, relu -> f32
//        2 = +bias -> f32
//        3 = +bias -> bf16 stored TRANSPOSED per batch: decT[(row>>6)*65536 + n*64 + (row&63)]
//        4 = +bias, sigmoid -> f32
//        5 = C[idx] = acc * C[idx]  (flow * gate fusion, in place)
template <int AMODE, int EMODE>
__global__ __launch_bounds__(NT) void splat_gemm(
    const float* __restrict__ Af, const uint16_t* __restrict__ Abf,
    long a_zoff, long lda,
    const uint16_t* __restrict__ Bt, long b_zoff, long ldb,   // Bt is [N][Kdim] bf16
    const float* __restrict__ scale, long s_zoff,
    const float* __restrict__ bias, long bias_zoff,
    float* __restrict__ Cf, uint16_t* __restrict__ Cbf, long c_zoff, long ldc,
    int M, int K)
{
  constexpr bool A_TDM = (AMODE == 1);

  __shared__ __align__(16) uint16_t sA[2][BM * BK];
  __shared__ __align__(16) uint16_t sB[2][BN * BK];

  const int tid  = threadIdx.x;
  const int lane = tid & 31;
  const int w    = tid >> 5;
  const int wm   = w >> 2;                    // 0..1 -> M offset wm*32
  const int wn   = w & 3;                     // 0..3 -> N offset wn*32
  const int z    = blockIdx.z;
  const int m0   = blockIdx.y * BM;
  const int n0   = blockIdx.x * BN;
  const long ab  = (long)z * a_zoff;
  const long bb  = (long)z * b_zoff;
  const long cb  = (long)z * c_zoff;

  const uint32_t ldsA0 = (uint32_t)(uintptr_t)(void*)&sA[0][0];
  const uint32_t ldsA1 = (uint32_t)(uintptr_t)(void*)&sA[1][0];
  const uint32_t ldsB0 = (uint32_t)(uintptr_t)(void*)&sB[0][0];
  const uint32_t ldsB1 = (uint32_t)(uintptr_t)(void*)&sB[1][0];

  auto stageA = [&](uint16_t* dst, int kc) {
    uint32_t* d32 = reinterpret_cast<uint32_t*>(dst);
    if constexpr (AMODE == 0) {
      for (int i = tid; i < BM * BK / 2; i += NT) {         // 1024 dwords
        int m = i >> 4, dk = (i & 15) << 1;
        int gm = m0 + m;
        float v0 = 0.f, v1 = 0.f;
        if (gm < M) {
          const float* p = &Af[ab + (long)gm * lda + kc + dk];
          v0 = p[0]; v1 = p[1];
          if constexpr (EMODE == 1) {
            float sc = scale[(long)z * s_zoff + gm];
            v0 *= sc; v1 *= sc;
          }
        }
        d32[m * (BK / 2) + (i & 15)] = (uint32_t)f2bf(v0) | ((uint32_t)f2bf(v1) << 16);
      }
    } else if constexpr (AMODE == 2) {  // tile columns are contiguous rows of src
      const uint32_t* s32 = reinterpret_cast<const uint32_t*>(Abf);
      for (int i = tid; i < BM * BK / 2; i += NT) {         // 1024 dwords
        int kk = i >> 5, dm = i & 31;                       // BM/2 = 32 dwords per k
        uint32_t v = s32[(ab + (long)(kc + kk) * lda) / 2 + dm];
        dst[(2 * dm) * BK + kk]     = (uint16_t)(v & 0xFFFFu);
        dst[(2 * dm + 1) * BK + kk] = (uint16_t)(v >> 16);
      }
    }
  };

  // Issue TDM DMAs for chunk kc into buffer `buf` (wave 0 only, uniform values).
  auto tdm_stage = [&](int buf, int kc) {
    if constexpr (A_TDM)
      tdm_load_2d(buf ? ldsA1 : ldsA0, Abf + ab + (long)m0 * lda + kc,
                  BK, BM, (uint32_t)lda);
    tdm_load_2d(buf ? ldsB1 : ldsB0, Bt + bb + (long)n0 * ldb + kc,
                BK, BN, (uint32_t)ldb);
  };

  v8f acc[2][2] = {};
  int cur = 0;
  if (w == 0) tdm_stage(0, 0);
  if constexpr (!A_TDM) stageA(sA[0], 0);

  for (int k0 = 0; k0 < K; k0 += BK) {
    if (w == 0) __builtin_amdgcn_s_wait_tensorcnt(0);
    __syncthreads();
    int nxt = cur ^ 1;
    if (k0 + BK < K) {
      if (w == 0) tdm_stage(nxt, k0 + BK);
      if constexpr (!A_TDM) stageA(sA[nxt], k0 + BK);
    }

    const uint16_t* tA = sA[cur];
    const uint16_t* tB = sB[cur];
    v16bf a0 = load_frag(tA + (wm * 32 + 0)  * BK, lane, true);
    v16bf a1 = load_frag(tA + (wm * 32 + 16) * BK, lane, true);
    v16bf b0 = load_frag(tB + (wn * 32 + 0)  * BK, lane, false);
    v16bf b1 = load_frag(tB + (wn * 32 + 16) * BK, lane, false);
    acc[0][0] = wmma_bf16(a0, b0, acc[0][0]);
    acc[0][1] = wmma_bf16(a0, b1, acc[0][1]);
    acc[1][0] = wmma_bf16(a1, b0, acc[1][0]);
    acc[1][1] = wmma_bf16(a1, b1, acc[1][1]);
    cur = nxt;
  }

  // epilogue: C/D layout — lane(n = L&15); VGPR j -> m = j + 8*(L>>4)
#pragma unroll
  for (int sm = 0; sm < 2; ++sm) {
    int rowbase = m0 + wm * 32 + sm * 16 + ((lane >> 4) << 3);
#pragma unroll
    for (int sn = 0; sn < 2; ++sn) {
      int n = n0 + wn * 32 + sn * 16 + (lane & 15);
      float bn = 0.f;
      if constexpr (EMODE >= 1 && EMODE <= 4) bn = bias[(long)z * bias_zoff + n];
#pragma unroll
      for (int j = 0; j < 8; ++j) {
        int row = rowbase + j;
        if (row >= M) continue;
        float v = acc[sm][sn][j];
        if constexpr (EMODE == 0) {
          Cf[cb + (long)row * ldc + n] = v;
        } else if constexpr (EMODE == 1) {
          v += bn; Cf[cb + (long)row * ldc + n] = v > 0.f ? v : 0.f;
        } else if constexpr (EMODE == 2) {
          Cf[cb + (long)row * ldc + n] = v + bn;
        } else if constexpr (EMODE == 3) {
          // store decoded^T per batch: [b][d][k]  (b = row>>6, k = row&63, d = n)
          Cbf[(long)(row >> 6) * 65536 + (long)n * 64 + (row & 63)] = f2bf(v + bn);
        } else if constexpr (EMODE == 4) {
          Cf[cb + (long)row * ldc + n] = 1.f / (1.f + __expf(-(v + bn)));
        } else {
          long cidx = cb + (long)row * ldc + n;
          Cf[cidx] = v * Cf[cidx];
        }
      }
    }
  }
}

// f32 -> bf16 elementwise, 4 per thread
__global__ void splat_convert4(const float* __restrict__ src, uint16_t* __restrict__ dst,
                               long n4) {
  long i = (long)blockIdx.x * blockDim.x + threadIdx.x;
  if (i >= n4) return;
  float4 f = reinterpret_cast<const float4*>(src)[i];
  union { uint16_t h[4]; uint2 u; } o;
  o.h[0] = f2bf(f.x); o.h[1] = f2bf(f.y); o.h[2] = f2bf(f.z); o.h[3] = f2bf(f.w);
  reinterpret_cast<uint2*>(dst)[i] = o.u;
}

// f32 [z][R][C] -> bf16 [z][C][R], LDS-tiled 32x32, block (32,8)
__global__ void splat_transpose_bf16(const float* __restrict__ src,
                                     uint16_t* __restrict__ dst, int R, int C) {
  __shared__ float tile[32][33];
  int z = blockIdx.z;
  long sb = (long)z * R * C;
  int c0 = blockIdx.x * 32, r0 = blockIdx.y * 32;
  int tx = threadIdx.x, ty = threadIdx.y;
  for (int yy = ty; yy < 32; yy += 8)
    tile[yy][tx] = src[sb + (long)(r0 + yy) * C + c0 + tx];
  __syncthreads();
  for (int yy = ty; yy < 32; yy += 8)
    dst[sb + (long)(c0 + yy) * R + r0 + tx] = f2bf(tile[tx][yy]);
}

// inv_tot[b*64+k] = 1 / max(sum_s infl[b,s,k], 1e-8)
__global__ void splat_inv_totals(const float* __restrict__ infl, float* __restrict__ inv_tot,
                                 int Bn, int Sn, int Kn) {
  int idx = blockIdx.x * blockDim.x + threadIdx.x;
  if (idx >= Bn * Kn) return;
  int b = idx / Kn, k = idx % Kn;
  const float* p = infl + (long)b * Sn * Kn + k;
  float s = 0.f;
  for (int t = 0; t < Sn; ++t) s += p[(long)t * Kn];
  if (s < 1e-8f) s = 1e-8f;
  inv_tot[idx] = 1.f / s;
}

// ---------------------------------------------------------------------------
extern "C" void kernel_launch(void* const* d_in, const int* in_sizes, int n_in,
                              void* d_out, int out_size, void* d_ws, size_t ws_size,
                              hipStream_t stream) {
  constexpr int B = 8, S = 2048, D = 1024, BD = 512, K = 64;

  const float* emb     = (const float*)d_in[0];  // [B,S,D]
  const float* infl    = (const float*)d_in[1];  // [B,S,K]
  const float* enc_W   = (const float*)d_in[2];  // [D,BD]
  const float* enc_b   = (const float*)d_in[3];  // [BD]
  const float* trans_W = (const float*)d_in[4];  // [K,BD,BD]
  const float* trans_b = (const float*)d_in[5];  // [K,BD]
  const float* dec_W   = (const float*)d_in[6];  // [BD,D]
  const float* dec_b   = (const float*)d_in[7];  // [D]
  const float* gate_W  = (const float*)d_in[8];  // [D,D]
  const float* gate_b  = (const float*)d_in[9];  // [D]
  float* out = (float*)d_out;                    // [B,S,D]
  (void)in_sizes; (void)n_in; (void)out_size; (void)ws_size;

  char* ws = (char*)d_ws;
  size_t off = 0;
  auto alloc = [&](size_t bytes) { char* p = ws + off; off = (off + bytes + 255) & ~(size_t)255; return p; };
  uint16_t* emb_bf   = (uint16_t*)alloc((size_t)B * S * D * 2);    // 32 MB, row-major (gate A)
  uint16_t* embT_bf  = (uint16_t*)alloc((size_t)B * D * S * 2);    // 32 MB, [b][d][s] (gather B^T)
  uint16_t* infl_bf  = (uint16_t*)alloc((size_t)B * S * K * 2);    //  2 MB
  uint16_t* encWT    = (uint16_t*)alloc((size_t)BD * D * 2);       //  1 MB  [BD][D]
  uint16_t* transWT  = (uint16_t*)alloc((size_t)K * BD * BD * 2);  // 32 MB  [k][e][d]
  uint16_t* decWT    = (uint16_t*)alloc((size_t)D * BD * 2);       //  1 MB  [D][BD]
  uint16_t* gateWT   = (uint16_t*)alloc((size_t)D * D * 2);        //  2 MB  [D][D]
  float*    inv_tot  = (float*)   alloc((size_t)B * K * 4);
  float*    G        = (float*)   alloc((size_t)B * K * D * 4);    //  2 MB gathered
  float*    E        = (float*)   alloc((size_t)B * K * BD * 4);   //  1 MB encoded
  float*    T        = (float*)   alloc((size_t)B * K * BD * 4);   //  1 MB transformed
  uint16_t* decT     = (uint16_t*)alloc((size_t)B * D * K * 2);    //  1 MB decoded^T [b][d][k]

  auto conv = [&](const float* s, uint16_t* d, long n) {
    long n4 = n / 4;
    splat_convert4<<<dim3((unsigned)((n4 + 255) / 256)), dim3(256), 0, stream>>>(s, d, n4);
  };
  auto convT = [&](const float* s, uint16_t* d, int R, int C, int Z) {
    splat_transpose_bf16<<<dim3(C / 32, R / 32, Z), dim3(32, 8), 0, stream>>>(s, d, R, C);
  };
  conv (emb,     emb_bf,  (long)B * S * D);
  conv (infl,    infl_bf, (long)B * S * K);
  convT(emb,     embT_bf, S, D, B);        // [b][d][s]
  convT(enc_W,   encWT,   D, BD, 1);       // [bd][d]
  convT(trans_W, transWT, BD, BD, K);      // [k][e][d]
  convT(dec_W,   decWT,   BD, D, 1);       // [d][bd]
  convT(gate_W,  gateWT,  D, D, 1);        // [d][d]

  splat_inv_totals<<<dim3(2), dim3(256), 0, stream>>>(infl, inv_tot, B, S, K);

  // 1) gather: G[b] = infl[b]^T (64xS) @ emb[b] (SxD);  A(m,k)=infl[s=k][m], B^T=embT
  splat_gemm<2, 0><<<dim3(D / BN, 1, B), dim3(NT), 0, stream>>>(
      nullptr, infl_bf, (long)S * K, K,
      embT_bf, (long)D * S, S,
      nullptr, 0, nullptr, 0,
      G, nullptr, (long)K * D, D, K, S);

  // 2) encode: E = relu((G * inv_tot) @ enc_W + enc_b)
  splat_gemm<0, 1><<<dim3(BD / BN, 1, B), dim3(NT), 0, stream>>>(
      G, nullptr, (long)K * D, D,
      encWT, 0, D,
      inv_tot, K, enc_b, 0,
      E, nullptr, (long)K * BD, BD, K, D);

  // 3) per-splat transform (z = splat, M = 8 batch rows, zero-padded tile)
  splat_gemm<0, 2><<<dim3(BD / BN, 1, K), dim3(NT), 0, stream>>>(
      E, nullptr, (long)BD, (long)K * BD,
      transWT, (long)BD * BD, BD,
      nullptr, 0, trans_b, BD,
      T, nullptr, (long)BD, (long)K * BD, B, BD);

  // 4) decode: all (b,k) rows flat, M=512; stores decoded^T bf16 (flow's B^T)
  splat_gemm<0, 3><<<dim3(D / BN, (B * K) / BM, 1), dim3(NT), 0, stream>>>(
      T, nullptr, 0, BD,
      decWT, 0, BD,
      nullptr, 0, dec_b, 0,
      nullptr, decT, 0, D, B * K, BD);

  // 5) gate: out = sigmoid(emb[b] @ gate_W + gate_b)
  splat_gemm<1, 4><<<dim3(D / BN, S / BM, B), dim3(NT), 0, stream>>>(
      nullptr, emb_bf, (long)S * D, D,
      gateWT, 0, D,
      nullptr, 0, gate_b, 0,
      out, nullptr, (long)S * D, D, S, D);

  // 6) flow & fuse: out = (infl[b] @ decoded[b]) * out
  splat_gemm<1, 5><<<dim3(D / BN, S / BM, B), dim3(NT), 0, stream>>>(
      nullptr, infl_bf, (long)S * K, K,
      decT, (long)D * K, K,
      nullptr, 0, nullptr, 0,
      out, nullptr, (long)S * D, D, S, K);
}
